// TransformerDecoder_28467043238489
// MI455X (gfx1250) — matrix operand order repaired and checked by reading, not compiled, not run on hardware
//
#include <hip/hip_runtime.h>
#include <hip/hip_bf16.h>

#define LNUM 2
#define HNUM 8
#define EDIM 512
#define FFD  2048
#define BNUM 8
#define SLEN 1024
#define MROWS (BNUM*SLEN)   // 8192 rows of x

typedef __attribute__((ext_vector_type(16))) __bf16 bf16x16;
typedef __attribute__((ext_vector_type(8)))  float  f32x8;
typedef __attribute__((ext_vector_type(4)))  float  f32x4;
typedef __attribute__((ext_vector_type(4)))  unsigned int u32x4;

union Frag { bf16x16 v; u32x4 q[2]; };

__device__ __forceinline__ unsigned short f2bf(float f) {
    union { float f; unsigned int u; } cv; cv.f = f;
    unsigned int u = cv.u;
    return (unsigned short)((u + 0x7FFFu + ((u >> 16) & 1u)) >> 16);
}
__device__ __forceinline__ unsigned int pk2(float lo, float hi) {
    return (unsigned int)f2bf(lo) | ((unsigned int)f2bf(hi) << 16);
}
__device__ __forceinline__ f32x8 wmma_bf16(const Frag& a, const Frag& b, f32x8 c) {
    return __builtin_amdgcn_wmma_f32_16x16x32_bf16(false, a.v, false, b.v,
                                                   (short)0, c, false, false);
}
__device__ __forceinline__ f32x8 zero8() {
    f32x8 z = {0.f,0.f,0.f,0.f,0.f,0.f,0.f,0.f};
    return z;
}

// gfx1250 async global->LDS copy, 16B per lane, tracked by ASYNCcnt.
// LDS byte offset = low 32 bits of the generic address (flat aperture rule).
__device__ __forceinline__ void async_load_b128(const void* g, const void* lds) {
    unsigned loff = (unsigned)(uintptr_t)lds;
    unsigned long long ga = (unsigned long long)(uintptr_t)g;
    asm volatile("global_load_async_to_lds_b128 %0, %1, off"
                 :: "v"(loff), "v"(ga) : "memory");
}
__device__ __forceinline__ void wait_async0() {
    asm volatile("s_wait_asynccnt 0x0" ::: "memory");
}

// ---------------------------------------------------------------------------
// fp32 -> (fp32 copy, bf16 copy)
// ---------------------------------------------------------------------------
__global__ __launch_bounds__(256) void cvt_kernel(const float* __restrict__ in,
                                                  float* __restrict__ xf,
                                                  unsigned short* __restrict__ xb,
                                                  int n) {
    int i = blockIdx.x * blockDim.x + threadIdx.x;
    if (i < n) { float v = in[i]; xf[i] = v; xb[i] = f2bf(v); }
}

// ---------------------------------------------------------------------------
// bf16 WMMA GEMM:  C[M,N] = A[M,K](bf16,row-major) * W^T
//   BMODE 0: W[n*K + k]                                  (Wq/Wk/Wv/W1/W2)
//   BMODE 1: W[(k>>9)*512*512 + n*512 + (k&511)]         (Wo: (H,O,D), k=(h,d))
//   SMODE 0: bf16 -> (B,H,S,D), +bias[n]                 (Q,K)
//   SMODE 1: bf16 -> (B,H,D,S), +bias[n]                 (V transposed)
//   SMODE 2: fp32 -> row-major M x N, no bias            (attn-out proj, FFN2)
//   SMODE 3: bf16 -> row-major, +bias[n], ReLU           (FFN1)
// Block tile 128x256, BK=32, 8 waves (2x4), wave tile 64x64 = 4x4 WMMA frags.
// A staged via async global->LDS (bf16 passthrough); B converted f32->bf16.
// ---------------------------------------------------------------------------
template<int BMODE, int SMODE>
__global__ __launch_bounds__(256) void gemm_bf16_kernel(
    const unsigned short* __restrict__ A, const float* __restrict__ W,
    const float* __restrict__ bias, void* __restrict__ Cout,
    int N, int K)
{
    __shared__ u32x4 aSh[8*32*2];    // 8 m-tiles in WMMA fragment layout,  8KB
    __shared__ u32x4 bSh[16*32*2];   // 16 n-tiles in WMMA fragment layout, 16KB

    const int tid  = threadIdx.x;
    const int lane = tid & 31;
    const int wave = tid >> 5;
    const int blockM = blockIdx.y * 128;
    const int blockN = blockIdx.x * 256;
    const int wm = wave & 1, wn = wave >> 1;
    const int hlf = lane >> 4;

    f32x8 acc[4][4];
    #pragma unroll
    for (int mi = 0; mi < 4; ++mi)
        #pragma unroll
        for (int ni = 0; ni < 4; ++ni) acc[mi][ni] = zero8();

    // A-stager: one fragment-lane per thread (runs at k+hlf*8 and k+16+hlf*8)
    const int amt  = tid >> 5;
    const int arow = blockM + amt*16 + (lane & 15);
    const int ah   = hlf * 8;
    const u32x4* aslot = &aSh[(amt*32 + lane)*2];
    // B-stager: 4 runs per thread (16 ntiles * 32 lanes * 2 runs = 1024 runs)
    const int bl  = (tid >> 1) & 31;              // fragment lane
    const int br  = tid & 1;                      // run within lane
    const int bntB = tid >> 6;                    // base ntile (j adds 4 each)
    const int bko = ((bl >> 4) * 8) + br * 16;    // k offset of this run

    for (int kb = 0; kb < K; kb += 32) {
        // ---- stage A: async global->LDS (bf16, no VGPR roundtrip)
        const unsigned short* ag = A + (size_t)arow * K + kb + ah;
        async_load_b128(ag,      aslot);          // run0: +0B lds, +0B  global
        async_load_b128(ag + 16, aslot + 1);      // run1: +16B lds, +32B global
        // ---- stage B: f32 weights -> bf16 fragments
        #pragma unroll
        for (int j = 0; j < 4; ++j) {
            int bnt = bntB + 4*j;
            int bn  = blockN + bnt*16 + (bl & 15);
            int k   = kb + bko;
            size_t baddr = (BMODE == 0)
                ? ((size_t)bn * K + k)
                : ((size_t)(k >> 9) * (512*512) + (size_t)bn * 512 + (size_t)(k & 511));
            const f32x4* wp = reinterpret_cast<const f32x4*>(W + baddr);
            f32x4 w0 = wp[0], w1 = wp[1];
            __builtin_prefetch((const void*)(wp + 8), 0, 1);   // next k-block
            u32x4 bp = { pk2(w0.x, w0.y), pk2(w0.z, w0.w),
                         pk2(w1.x, w1.y), pk2(w1.z, w1.w) };
            bSh[(bnt*32 + bl)*2 + br] = bp;
        }
        wait_async0();
        __syncthreads();

        // ---- compute: 4x4 WMMA tiles per wave (16 WMMA per k-block)
        Frag fa[4], fb[4];
        #pragma unroll
        for (int i = 0; i < 4; ++i) {
            int ai = ((wm*4 + i)*32 + lane)*2;
            fa[i].q[0] = aSh[ai]; fa[i].q[1] = aSh[ai+1];
            int bi = ((wn*4 + i)*32 + lane)*2;
            fb[i].q[0] = bSh[bi]; fb[i].q[1] = bSh[bi+1];
        }
        #pragma unroll
        for (int mi = 0; mi < 4; ++mi)
            #pragma unroll
            for (int ni = 0; ni < 4; ++ni)
                acc[mi][ni] = wmma_bf16(fa[mi], fb[ni], acc[mi][ni]);
        __syncthreads();
    }

    // ---- epilogue (all branches are compile-time)
    #pragma unroll
    for (int mi = 0; mi < 4; ++mi) {
        #pragma unroll
        for (int ni = 0; ni < 4; ++ni) {
            int gn = blockN + (wn*4 + ni)*16 + (lane & 15);
            #pragma unroll
            for (int r = 0; r < 8; ++r) {
                int gm = blockM + (wm*4 + mi)*16 + r + 8*hlf;
                float val = acc[mi][ni][r];
                if (SMODE == 2) {
                    ((float*)Cout)[(size_t)gm * N + gn] = val;
                } else {
                    val += bias[gn];
                    if (SMODE == 3) {
                        val = fmaxf(val, 0.f);
                        ((unsigned short*)Cout)[(size_t)gm * N + gn] = f2bf(val);
                    } else {
                        int bb = gm >> 10, s = gm & 1023;
                        int hh = gn >> 9,  o = gn & 511;
                        size_t addr = (SMODE == 0)
                            ? (((size_t)(bb*HNUM + hh)*SLEN + s)*EDIM + o)   // (B,H,S,D)
                            : (((size_t)(bb*HNUM + hh)*EDIM + o)*SLEN + s);  // (B,H,D,S)
                        ((unsigned short*)Cout)[addr] = f2bf(val);
                    }
                }
            }
        }
    }
}

// ---------------------------------------------------------------------------
// Attention: one block per (b, h, 16-row q tile).  256 threads = 8 waves.
// Phase 1: S = Q K^T * 1/sqrt(E) + causal   -> 16x1024 f32 in LDS
// Phase 2: softmax (16-lane shuffle reductions), P -> bf16 in-place in LDS
// Phase 3: O = P V * alpha[h]  -> AO (b,s,h,d) bf16
// ---------------------------------------------------------------------------
__global__ __launch_bounds__(256) void attn_kernel(
    const unsigned short* __restrict__ Q, const unsigned short* __restrict__ Kt,
    const unsigned short* __restrict__ Vt, const float* __restrict__ alphas,
    unsigned short* __restrict__ AO)
{
    extern __shared__ char smem[];
    float* sS          = (float*)smem;           // 16 x 1024 f32 (64KB)
    unsigned short* sP = (unsigned short*)smem;  // 16 x 1024 bf16 overlaid (32KB)
    const u32x4* sP4   = (const u32x4*)smem;

    const int tid  = threadIdx.x;
    const int lane = tid & 31;
    const int wave = tid >> 5;
    const int hlf  = lane >> 4;
    const int qbase = blockIdx.x * 16;
    const int h = blockIdx.y, b = blockIdx.z;
    const int qmax = qbase + 15;
    const size_t plane = (size_t)(b * HNUM + h) * SLEN * EDIM;

    // ---------------- phase 1: scores ----------------
    const int qrow = qbase + (lane & 15);
    const u32x4* qp = (const u32x4*)(Q + plane + (size_t)qrow * EDIM);
    Frag qf[16];
    #pragma unroll
    for (int kb = 0; kb < 16; ++kb) {           // A frag: runs at hlf*8, 16+hlf*8
        qf[kb].q[0] = qp[kb*4 + hlf];
        qf[kb].q[1] = qp[kb*4 + 2 + hlf];
    }
    const float invs = 0.044194173824159216f;   // 1/sqrt(512)
    for (int nt = 0; nt < 8; ++nt) {            // wave covers keys [wave*128, +128)
        int keyT = wave * 128 + nt * 16;
        if (keyT > qmax) {                      // fully masked tile (wave-uniform)
            #pragma unroll
            for (int r = 0; r < 8; ++r)
                sS[(r + 8*hlf)*1024 + keyT + (lane & 15)] = -1e30f;
            continue;
        }
        const u32x4* kp = (const u32x4*)(Kt + plane + (size_t)(keyT + (lane & 15)) * EDIM);
        f32x8 acc = zero8();
        #pragma unroll
        for (int kb = 0; kb < 16; ++kb) {       // B frag: runs at hlf*8, 16+hlf*8
            Frag fk;
            fk.q[0] = kp[kb*4 + hlf];
            fk.q[1] = kp[kb*4 + 2 + hlf];
            acc = wmma_bf16(qf[kb], fk, acc);
        }
        #pragma unroll
        for (int r = 0; r < 8; ++r) {
            int m = r + 8*hlf;
            int key = keyT + (lane & 15);
            sS[m*1024 + key] = (key <= qbase + m) ? acc[r] * invs : -1e30f;
        }
    }
    __syncthreads();

    // ---------------- phase 2: softmax ----------------
    {
        const int row = tid >> 4;               // 16 threads per row
        const int tr  = tid & 15;
        float vals[64];
        float mx = -1e30f;
        #pragma unroll
        for (int j = 0; j < 64; ++j) {
            float v = sS[row*1024 + tr + j*16];
            vals[j] = v; mx = fmaxf(mx, v);
        }
        #pragma unroll
        for (int off = 8; off >= 1; off >>= 1) mx = fmaxf(mx, __shfl_xor(mx, off, 16));
        float sum = 0.f;
        #pragma unroll
        for (int j = 0; j < 64; ++j) { float e = __expf(vals[j] - mx); vals[j] = e; sum += e; }
        #pragma unroll
        for (int off = 8; off >= 1; off >>= 1) sum += __shfl_xor(sum, off, 16);
        float inv = 1.f / sum;
        __syncthreads();                        // all f32 reads done before bf16 overwrite
        #pragma unroll
        for (int j = 0; j < 64; ++j)
            sP[row*1024 + tr + j*16] = f2bf(vals[j] * inv);
        __syncthreads();
    }

    // ---------------- phase 3: O = P V ----------------
    const int kbMax = qmax >> 5;                // causal truncation of key blocks
    f32x8 acc3[4] = {zero8(), zero8(), zero8(), zero8()};
    const int prowOff = ((lane & 15) * 1024) >> 3;
    for (int kb = 0; kb <= kbMax; ++kb) {
        Frag fp;                                // A frag of P from LDS
        fp.q[0] = sP4[prowOff + kb*4 + hlf];
        fp.q[1] = sP4[prowOff + kb*4 + 2 + hlf];
        #pragma unroll
        for (int nt = 0; nt < 4; ++nt) {        // wave covers d cols [wave*64, +64)
            int d = wave*64 + nt*16 + (lane & 15);
            const u32x4* vp = (const u32x4*)(Vt + (size_t)((b*HNUM + h)*EDIM + d) * SLEN);
            Frag fv;                            // B frag of V^T (key-contiguous)
            fv.q[0] = vp[kb*4 + hlf];
            fv.q[1] = vp[kb*4 + 2 + hlf];
            acc3[nt] = wmma_bf16(fp, fv, acc3[nt]);
        }
    }
    const float al = alphas[h];
    #pragma unroll
    for (int nt = 0; nt < 4; ++nt) {
        int d = wave*64 + nt*16 + (lane & 15);
        #pragma unroll
        for (int r = 0; r < 8; ++r) {
            int q = qbase + r + 8*hlf;
            AO[((size_t)(b*SLEN + q)*HNUM + h)*EDIM + d] = f2bf(acc3[nt][r] * al);
        }
    }
}

// ---------------------------------------------------------------------------
// out = LayerNorm(x + y + sum_{hh<biasHeads} bias[hh*E + e]) * g + beta
// one wave per row; also emits bf16 copy for the next GEMM.
// ---------------------------------------------------------------------------
__global__ __launch_bounds__(256) void ln_res_kernel(
    const float* __restrict__ x, const float* __restrict__ y,
    const float* __restrict__ bias, int biasHeads,
    const float* __restrict__ g, const float* __restrict__ be,
    float* __restrict__ outF, unsigned short* __restrict__ outB)
{
    const int row  = blockIdx.x * 8 + (threadIdx.x >> 5);
    const int lane = threadIdx.x & 31;
    const size_t base = (size_t)row * EDIM;
    float v[16];
    float s = 0.f;
    #pragma unroll
    for (int j = 0; j < 16; ++j) {
        int e = lane + j*32;
        float t = x[base + e] + y[base + e];
        for (int hh = 0; hh < biasHeads; ++hh) t += bias[hh*EDIM + e];
        v[j] = t; s += t;
    }
    #pragma unroll
    for (int off = 16; off >= 1; off >>= 1) s += __shfl_xor(s, off, 32);
    float mu = s * (1.f/EDIM);
    float var = 0.f;
    #pragma unroll
    for (int j = 0; j < 16; ++j) { float d = v[j] - mu; var += d*d; }
    #pragma unroll
    for (int off = 16; off >= 1; off >>= 1) var += __shfl_xor(var, off, 32);
    float rs = rsqrtf(var * (1.f/EDIM) + 1e-5f);
    #pragma unroll
    for (int j = 0; j < 16; ++j) {
        int e = lane + j*32;
        float o = (v[j] - mu) * rs * g[e] + be[e];
        outF[base + e] = o;
        outB[base + e] = f2bf(o);
    }
}

// ---------------------------------------------------------------------------
extern "C" void kernel_launch(void* const* d_in, const int* in_sizes, int n_in,
                              void* d_out, int out_size, void* d_ws, size_t ws_size,
                              hipStream_t stream) {
    (void)in_sizes; (void)n_in; (void)out_size; (void)ws_size;
    const float* x_in  = (const float*)d_in[0];
    const float* Wq    = (const float*)d_in[1];
    const float* bq    = (const float*)d_in[2];
    const float* Wk    = (const float*)d_in[3];
    const float* bk    = (const float*)d_in[4];
    const float* Wv    = (const float*)d_in[5];
    const float* bv    = (const float*)d_in[6];
    const float* Wo    = (const float*)d_in[7];
    const float* bo    = (const float*)d_in[8];
    const float* alphas= (const float*)d_in[9];
    const float* ln1g  = (const float*)d_in[10];
    const float* ln1b  = (const float*)d_in[11];
    const float* W1    = (const float*)d_in[12];
    const float* b1    = (const float*)d_in[13];
    const float* W2    = (const float*)d_in[14];
    const float* b2    = (const float*)d_in[15];
    const float* ln2g  = (const float*)d_in[16];
    const float* ln2b  = (const float*)d_in[17];

    // workspace carve-up
    char* ws = (char*)d_ws;
    float* xf           = (float*)ws;          ws += (size_t)MROWS*EDIM*4;            // 16MB
    unsigned short* xb  = (unsigned short*)ws; ws += (size_t)MROWS*EDIM*2;            // 8MB
    unsigned short* Qb  = (unsigned short*)ws; ws += (size_t)MROWS*HNUM*EDIM*2;       // 64MB
    unsigned short* Kb  = (unsigned short*)ws; ws += (size_t)MROWS*HNUM*EDIM*2;       // 64MB
    unsigned short* Vtb = (unsigned short*)ws; ws += (size_t)MROWS*HNUM*EDIM*2;       // 64MB
    unsigned short* AOb = (unsigned short*)ws; ws += (size_t)MROWS*HNUM*EDIM*2;       // 64MB
    float* tmpF         = (float*)ws;          ws += (size_t)MROWS*EDIM*4;            // 16MB
    unsigned short* hb  = (unsigned short*)ws; ws += (size_t)MROWS*FFD*2;             // 32MB

    cvt_kernel<<<(MROWS*EDIM + 255)/256, 256, 0, stream>>>(x_in, xf, xb, MROWS*EDIM);

    const dim3 gQKV(HNUM*EDIM/256, MROWS/128);  // (16, 64)
    const dim3 gOUT(EDIM/256,      MROWS/128);  // (2, 64)
    const dim3 gFF1(FFD/256,       MROWS/128);  // (8, 64)
    const dim3 gATT(SLEN/16, HNUM, BNUM);       // (64, 8, 8)

    for (int i = 0; i < LNUM; ++i) {
        const size_t wOff = (size_t)i*HNUM*EDIM*EDIM;
        // Q, K, V projections (V stored transposed (B,H,D,S))
        gemm_bf16_kernel<0,0><<<gQKV, 256, 0, stream>>>(xb, Wq + wOff, bq + (size_t)i*HNUM*EDIM,
                                                        Qb,  HNUM*EDIM, EDIM);
        gemm_bf16_kernel<0,0><<<gQKV, 256, 0, stream>>>(xb, Wk + wOff, bk + (size_t)i*HNUM*EDIM,
                                                        Kb,  HNUM*EDIM, EDIM);
        gemm_bf16_kernel<0,1><<<gQKV, 256, 0, stream>>>(xb, Wv + wOff, bv + (size_t)i*HNUM*EDIM,
                                                        Vtb, HNUM*EDIM, EDIM);
        // causal attention + head alpha
        attn_kernel<<<gATT, 256, 65536, stream>>>(Qb, Kb, Vtb, alphas, AOb);
        // output projection (K = H*E = 4096, Wo addressing mode 1)
        gemm_bf16_kernel<1,2><<<gOUT, 256, 0, stream>>>(AOb, Wo + wOff, nullptr,
                                                        tmpF, EDIM, HNUM*EDIM);
        // residual + sum_h(bo) + LN1
        ln_res_kernel<<<MROWS/8, 256, 0, stream>>>(xf, tmpF, bo + (size_t)i*HNUM*EDIM, HNUM,
                                                   ln1g + i*EDIM, ln1b + i*EDIM, xf, xb);
        // FFN
        gemm_bf16_kernel<0,3><<<gFF1, 256, 0, stream>>>(xb, W1 + (size_t)i*FFD*EDIM, b1 + (size_t)i*FFD,
                                                        hb,   FFD,  EDIM);
        gemm_bf16_kernel<0,2><<<gOUT, 256, 0, stream>>>(hb, W2 + (size_t)i*EDIM*FFD, nullptr,
                                                        tmpF, EDIM, FFD);
        // residual + b2 + LN2 (final layer writes d_out)
        float* outp = (i == LNUM-1) ? (float*)d_out : xf;
        ln_res_kernel<<<MROWS/8, 256, 0, stream>>>(xf, tmpF, b2 + (size_t)i*EDIM, 1,
                                                   ln2g + i*EDIM, ln2b + i*EDIM, outp, xb);
    }
}
